// ResonanceLayer_41532333752643
// MI455X (gfx1250) — matrix operand order repaired and checked by reading, not compiled, not run on hardware
//
#include <hip/hip_runtime.h>
#include <cstdint>
#include <cstddef>

// Problem constants (match reference)
#define B_CNT  64
#define N_CNT  128
#define T_CNT  32
#define DH     128
#define DHALF  64
#define P_CNT  8
#define TWO_PI_F 6.283185307179586f

// LDS strides (elements). 136*2B = 272B rows: 16B-aligned for ds_load_b128,
// and 272/4 = 68 == 4 (mod 64 banks) breaks worst-case conflicts.
#define LDA 136
#define LDW 136

typedef __attribute__((ext_vector_type(16))) __bf16 v16bf;
typedef __attribute__((ext_vector_type(8)))  float  v8f;

union FragAB { v16bf v; uint4 q[2]; };

__device__ __forceinline__ unsigned short f32_to_bf16_rne(float f) {
  unsigned int u = __float_as_uint(f);
  unsigned int r = u + 0x7FFFu + ((u >> 16) & 1u);   // round-to-nearest-even
  return (unsigned short)(r >> 16);
}

// One 128xK @ KxNT*16 GEMM for this wave's 16-row strip.
// A tile:  ldsA, rows r (stride LDA), bf16, K = 128
// B tile:  ldsW, stored TRANSPOSED: element (k, n) at ldsW[n*LDW + k]
// acc[nt]: f32 16x16 accumulators (v8f), per ISA C/D layout.
template<int NT>
__device__ __forceinline__ void gemm_strip(const unsigned short* ldsA_,
                                           const unsigned short* ldsW_,
                                           int ws, int lane, v8f* acc) {
  const int hf = lane >> 4;     // half-wave select
  const int m  = lane & 15;     // A: row-in-strip; B: output column-in-tile
  v8f z;
#pragma unroll
  for (int i = 0; i < 8; ++i) z[i] = 0.0f;
#pragma unroll
  for (int nt = 0; nt < NT; ++nt) acc[nt] = z;

#pragma unroll
  for (int kk = 0; kk < 4; ++kk) {            // K = 4 * 32
    // A fragment 16x32 bf16 (ISA 7.12.2): lane<16 -> elems 0..7 = K0..7,
    // elems 8..15 = K16..23 ; lane>=16 -> K8..15 / K24..31.
    FragAB a;
    const unsigned short* ap = ldsA_ + (ws * 16 + m) * LDA + kk * 32 + hf * 8;
    a.q[0] = *(const uint4*)(ap);
    a.q[1] = *(const uint4*)(ap + 16);
#pragma unroll
    for (int nt = 0; nt < NT; ++nt) {
      // B fragment 32x16 bf16: lanes 0..15 hold K0..7 / K16..23, lanes 16..31 K8..15 / K24..31,
      // N = lane&15. ldsW transposed -> 2x b128 of 8 contiguous K per lane.
      FragAB bfr;
      const unsigned short* bp = ldsW_ + (nt * 16 + m) * LDW + kk * 32 + hf * 8;
      bfr.q[0] = *(const uint4*)(bp);
      bfr.q[1] = *(const uint4*)(bp + 16);
      acc[nt] = __builtin_amdgcn_wmma_f32_16x16x32_bf16(
          false, a.v, false, bfr.v, (short)0, acc[nt], false, false);
    }
  }
}

// Dynamic-LDS layout sizes
#define SMEM_USHORTS (128*LDA + 128*LDW + 128*LDW + 64*LDW)           // A,W1,W2,W3
#define SMEM_FLOATS  (32*128 + 256 + 128 + 128 + 128 + 64 + 256 + 64) // ego,Wte,bte,b1,b2,b3,np,ep
#define SMEM_A_BYTES ((size_t)SMEM_USHORTS*2 + (size_t)SMEM_FLOATS*4) // ~139 KB (< 320 KB/WGP)

// ---------------------------------------------------------------------------
// Kernel A: fused TrajEncoding + ego*nei product + 3-layer WMMA MLP -> f_re
// grid = B * (N/4); block = 256 (8 wave32); each block: 4 neighbors x 32 t
// ---------------------------------------------------------------------------
__global__ __launch_bounds__(256) void resonance_mlp_kernel(
    const float* __restrict__ x_ego, const float* __restrict__ x_nei,
    const float* __restrict__ W_te, const float* __restrict__ b_te,
    const float* __restrict__ W1,   const float* __restrict__ b1,
    const float* __restrict__ W2,   const float* __restrict__ b2,
    const float* __restrict__ W3,   const float* __restrict__ b3,
    float* __restrict__ f_re_out) {
  extern __shared__ unsigned char smem_raw[];
  unsigned short* ldsA  = (unsigned short*)smem_raw;     // 128 x LDA bf16 activations
  unsigned short* ldsW1 = ldsA  + 128 * LDA;             // W1^T  bf16
  unsigned short* ldsW2 = ldsW1 + 128 * LDW;             // W2^T  bf16
  unsigned short* ldsW3 = ldsW2 + 128 * LDW;             // W3^T  bf16 (64 rows)
  float* ldsEgo = (float*)(ldsW3 + 64 * LDW);            // 32 x 128 f32 ego features
  float* ldsWte = ldsEgo + 32 * 128;                     // 2 x 128
  float* ldsBte = ldsWte + 256;                          // 128
  float* ldsB1  = ldsBte + 128;                          // 128
  float* ldsB2  = ldsB1  + 128;                          // 128
  float* ldsB3  = ldsB2  + 128;                          // 64
  float* ldsNP  = ldsB3  + 64;                           // 128 x 2 nei pure coords
  float* ldsEP  = ldsNP  + 256;                          // 32 x 2 ego pure coords

  const int tid  = threadIdx.x;
  const int lane = tid & 31;
  const int ws   = tid >> 5;                 // wave index -> 16-row strip
  const int b    = blockIdx.x >> 5;          // 32 neighbor-chunks per batch
  const int n0   = (blockIdx.x & 31) * 4;    // first neighbor of this chunk

  // ---- stage weights (transposed, bf16) + biases into LDS ----
  for (int idx = tid; idx < 128 * 128; idx += 256) {
    int k = idx >> 7, n = idx & 127;
    ldsW1[n * LDW + k] = f32_to_bf16_rne(W1[idx]);
    ldsW2[n * LDW + k] = f32_to_bf16_rne(W2[idx]);
  }
  for (int idx = tid; idx < 128 * 64; idx += 256) {
    int k = idx >> 6, n = idx & 63;
    ldsW3[n * LDW + k] = f32_to_bf16_rne(W3[idx]);
  }
  ldsWte[tid] = W_te[tid];                   // 256 == blockDim
  if (tid < 128) { ldsBte[tid] = b_te[tid]; ldsB1[tid] = b1[tid]; ldsB2[tid] = b2[tid]; }
  if (tid < 64)  { ldsB3[tid] = b3[tid]; }

  // ---- pure (last-step-relative) coordinates ----
  if (tid < 128) {                            // row r -> (nl = r>>5, t = r&31)
    int nl = tid >> 5, t = tid & 31;
    size_t nb = (((size_t)b * N_CNT + (n0 + nl)) * T_CNT) * 2;
    float lx = x_nei[nb + 31 * 2 + 0], ly = x_nei[nb + 31 * 2 + 1];
    ldsNP[tid * 2 + 0] = x_nei[nb + t * 2 + 0] - lx;
    ldsNP[tid * 2 + 1] = x_nei[nb + t * 2 + 1] - ly;
  }
  if (tid < 32) {
    size_t eb = (size_t)b * T_CNT * 2;
    float lx = x_ego[eb + 31 * 2 + 0], ly = x_ego[eb + 31 * 2 + 1];
    ldsEP[tid * 2 + 0] = x_ego[eb + tid * 2 + 0] - lx;
    ldsEP[tid * 2 + 1] = x_ego[eb + tid * 2 + 1] - ly;
  }
  __syncthreads();

  // ---- f_ego = relu(ego_pure @ W_te + b_te)  [32 x 128] ----
  for (int idx = tid; idx < 32 * 128; idx += 256) {
    int t = idx >> 7, d = idx & 127;
    float v = fmaf(ldsEP[t * 2], ldsWte[d],
              fmaf(ldsEP[t * 2 + 1], ldsWte[128 + d], ldsBte[d]));
    ldsEgo[idx] = fmaxf(v, 0.0f);
  }
  __syncthreads();

  // ---- A = bf16( relu(nei_pure @ W_te + b_te) * f_ego )  [128 x 128] ----
  for (int idx = tid; idx < 128 * 128; idx += 256) {
    int r = idx >> 7, d = idx & 127;
    int t = r & 31;
    float v = fmaf(ldsNP[r * 2], ldsWte[d],
              fmaf(ldsNP[r * 2 + 1], ldsWte[128 + d], ldsBte[d]));
    v = fmaxf(v, 0.0f) * ldsEgo[t * 128 + d];
    ldsA[r * LDA + d] = f32_to_bf16_rne(v);
  }
  __syncthreads();

  const int hf = lane >> 4;
  const int m  = lane & 15;
  v8f acc[8];

  // ---- layer 1: H1 = relu(A @ W1 + b1) -> back into ldsA (bf16) ----
  gemm_strip<8>(ldsA, ldsW1, ws, lane, acc);
  __syncthreads();
#pragma unroll
  for (int nt = 0; nt < 8; ++nt)
#pragma unroll
    for (int v = 0; v < 8; ++v) {
      int row = ws * 16 + v + hf * 8;
      int col = nt * 16 + m;
      ldsA[row * LDA + col] = f32_to_bf16_rne(fmaxf(acc[nt][v] + ldsB1[col], 0.0f));
    }
  __syncthreads();

  // ---- layer 2: H2 = relu(H1 @ W2 + b2) -> back into ldsA (bf16) ----
  gemm_strip<8>(ldsA, ldsW2, ws, lane, acc);
  __syncthreads();
#pragma unroll
  for (int nt = 0; nt < 8; ++nt)
#pragma unroll
    for (int v = 0; v < 8; ++v) {
      int row = ws * 16 + v + hf * 8;
      int col = nt * 16 + m;
      ldsA[row * LDA + col] = f32_to_bf16_rne(fmaxf(acc[nt][v] + ldsB2[col], 0.0f));
    }
  __syncthreads();

  // ---- layer 3: f_re = relu(H2 @ W3 + b3) -> global [B,N,T,64] ----
  gemm_strip<4>(ldsA, ldsW3, ws, lane, acc);
#pragma unroll
  for (int nt = 0; nt < 4; ++nt)
#pragma unroll
    for (int v = 0; v < 8; ++v) {
      int row = ws * 16 + v + hf * 8;
      int col = nt * 16 + m;
      int nl = row >> 5, t = row & 31;
      size_t gi = (((size_t)(b * N_CNT + n0 + nl)) * T_CNT + t) * DHALF + col;
      f_re_out[gi] = fmaxf(acc[nt][v] + ldsB3[col], 0.0f);
    }
}

// ---------------------------------------------------------------------------
// Kernel B: angle-partition masked means over N (LDS ds_add_f32 reduction)
// + f_pos dense + re_matrix assembly.  grid = B*T; block = 256.
// ---------------------------------------------------------------------------
__global__ __launch_bounds__(256) void resonance_part_kernel(
    const float* __restrict__ x_ego, const float* __restrict__ x_nei,
    const float* __restrict__ W_ce,  const float* __restrict__ b_ce,
    const float* __restrict__ f_re,  float* __restrict__ re_out) {
  __shared__ float sCnt[P_CNT], sD[P_CNT], sA[P_CNT];
  __shared__ float sF[P_CNT * DHALF];
  __shared__ float sWce[2 * DHALF], sBce[DHALF];

  const int tid = threadIdx.x;
  const int b = blockIdx.x >> 5;
  const int t = blockIdx.x & 31;

  if (tid < P_CNT) { sCnt[tid] = 0.0f; sD[tid] = 0.0f; sA[tid] = 0.0f; }
  for (int i = tid; i < P_CNT * DHALF; i += 256) sF[i] = 0.0f;
  if (tid < 128) sWce[tid] = W_ce[tid];
  if (tid < 64)  sBce[tid] = b_ce[tid];
  __syncthreads();

  const size_t eb = (size_t)b * T_CNT * 2;
  const float ex  = x_ego[eb + t * 2 + 0], ey  = x_ego[eb + t * 2 + 1];
  const float elx = x_ego[eb + 31 * 2 + 0], ely = x_ego[eb + 31 * 2 + 1];

  // two threads per neighbor: hd selects which 32-wide half of f_re to reduce
  const int n  = tid >> 1;
  const int hd = tid & 1;
  {
    size_t nb = (((size_t)b * N_CNT + n) * T_CNT + t) * 2;
    float px = x_nei[nb + 0] + elx - ex;
    float py = x_nei[nb + 1] + ely - ey;
    float dist = sqrtf(px * px + py * py);
    float ang  = atan2f(px, py);                     // jnp.arctan2(p0, p1)
    ang = fmodf(ang, TWO_PI_F);
    if (ang < 0.0f) ang += TWO_PI_F;
    int part = (int)(ang * (P_CNT / TWO_PI_F));
    part &= (P_CNT - 1);
    bool valid = ((fabsf(px) + fabsf(py)) > 0.05f) && (dist > 0.005f);
    if (valid) {
      if (hd == 0) {
        atomicAdd(&sCnt[part], 1.0f);
        atomicAdd(&sD[part], dist);
        atomicAdd(&sA[part], ang);
      }
      const float* fr = f_re + (((size_t)(b * N_CNT + n)) * T_CNT + t) * DHALF + hd * 32;
      float* dst = &sF[part * DHALF + hd * 32];
#pragma unroll 8
      for (int d = 0; d < 32; ++d) atomicAdd(&dst[d], fr[d]);
    }
  }
  __syncthreads();

  // re_matrix[b,t,p,:] = [ re_partitions(64) | relu(positions @ W_ce + b_ce)(64) ]
  for (int idx = tid; idx < P_CNT * 128; idx += 256) {
    int p = idx >> 7, c = idx & 127;
    float denom = sCnt[p] + 1e-4f;
    float val;
    if (c < DHALF) {
      val = sF[p * DHALF + c] / denom;
    } else {
      int j = c - DHALF;
      float pd = sD[p] / denom;
      float pa = sA[p] / denom;
      val = fmaxf(fmaf(pd, sWce[j], fmaf(pa, sWce[DHALF + j], sBce[j])), 0.0f);
    }
    re_out[(((size_t)b * T_CNT + t) * P_CNT + p) * 128 + c] = val;
  }
}

// ---------------------------------------------------------------------------
extern "C" void kernel_launch(void* const* d_in, const int* in_sizes, int n_in,
                              void* d_out, int out_size, void* d_ws, size_t ws_size,
                              hipStream_t stream) {
  const float* x_ego = (const float*)d_in[0];
  const float* x_nei = (const float*)d_in[1];
  const float* W_te  = (const float*)d_in[2];
  const float* b_te  = (const float*)d_in[3];
  const float* W1    = (const float*)d_in[4];
  const float* b1    = (const float*)d_in[5];
  const float* W2    = (const float*)d_in[6];
  const float* b2    = (const float*)d_in[7];
  const float* W3    = (const float*)d_in[8];
  const float* b3    = (const float*)d_in[9];
  const float* W_ce  = (const float*)d_in[10];
  const float* b_ce  = (const float*)d_in[11];

  float* out       = (float*)d_out;
  float* re_matrix = out;                                   // [B,T,P,128]
  float* f_re      = out + (size_t)B_CNT * T_CNT * P_CNT * 128;  // [B,N,T,64]

  (void)in_sizes; (void)n_in; (void)out_size; (void)d_ws; (void)ws_size;

  const size_t shA = SMEM_A_BYTES;   // ~139 KB dynamic LDS (320 KB/WGP available)
  hipFuncSetAttribute(reinterpret_cast<const void*>(resonance_mlp_kernel),
                      hipFuncAttributeMaxDynamicSharedMemorySize, (int)shA);

  resonance_mlp_kernel<<<B_CNT * (N_CNT / 4), 256, shA, stream>>>(
      x_ego, x_nei, W_te, b_te, W1, b1, W2, b2, W3, b3, f_re);

  resonance_part_kernel<<<B_CNT * T_CNT, 256, 0, stream>>>(
      x_ego, x_nei, W_ce, b_ce, f_re, re_matrix);
}
